// EncoderNoResidual_55353538511032
// MI455X (gfx1250) — matrix operand order repaired and checked by reading, not compiled, not run on hardware
//
#include <hip/hip_runtime.h>
#include <math.h>

// ---------------------------------------------------------------------------
// CDNA5 (gfx1250) WMMA types
// ---------------------------------------------------------------------------
typedef __attribute__((ext_vector_type(16))) __bf16 v16bf;
typedef __attribute__((ext_vector_type(2)))  __bf16 v2bf;
typedef __attribute__((ext_vector_type(8)))  float  v8f;

union FragAB {
    v16bf v;
    unsigned int u32[8];
};

// Pack two f32 into two bf16 (low ushort = a, high ushort = b).
__device__ __forceinline__ unsigned int pack2(float a, float b) {
#if __has_builtin(__builtin_amdgcn_cvt_pk_bf16_f32)
    // native packed convert (round-to-nearest-even), one instruction
    v2bf r = __builtin_amdgcn_cvt_pk_bf16_f32(a, b);
    return __builtin_bit_cast(unsigned int, r);
#else
    // single v_perm_b32: select high 16 bits of each f32 (truncate-to-bf16)
    return __builtin_amdgcn_perm(__float_as_uint(b), __float_as_uint(a),
                                 0x07060302u);
#endif
}

// Tile geometry: 256 threads = 8 waves (wave32). Block tile 64(M) x 64(N),
// k-step 32. Waves arranged 4(M) x 2(N); each wave owns a 16x32 output strip:
// one A fragment + two B fragments -> 2 v_wmma_f32_16x16x32_bf16 per k-step.
// LDS tiles are double-buffered: one barrier per k-step, next tile's global
// loads issue before the current tile's WMMAs (latency hiding).
#define TM 64
#define TN 64
#define TK 32
#define AS 34   // LDS row stride (ushorts) for A tile: 17 banks -> conflict free
#define BS 66   // LDS row stride (ushorts) for B tile: 33 banks -> conflict free

// ---------------------------------------------------------------------------
// Generic bf16-WMMA GEMM:  C[M,N] = op(A[M,K] @ B[K,N] (+ bias))
// A,B,C f32 row-major (lda=K, ldb=ldc=N). M % 64 == 0 assumed.
// Optional C2 gets relu(result) (for the z / emb=relu(z) dual write).
// ---------------------------------------------------------------------------
__global__ __launch_bounds__(256) void gemm_bf16_wmma(
    const float* __restrict__ A, const float* __restrict__ B,
    const float* __restrict__ bias, float* __restrict__ C,
    float* __restrict__ C2, int M, int N, int K, int doRelu)
{
    __shared__ unsigned short Asm[2][TM * AS];
    __shared__ unsigned short Bsm[2][TK * BS];

    const int tid  = threadIdx.x;
    const int lane = tid & 31;
    const int w    = tid >> 5;
    const int wm   = w >> 1;          // 0..3
    const int wn   = w & 1;           // 0..1
    const int mBase = blockIdx.x * TM;
    const int nBase = blockIdx.y * TN;

    const int ar  = tid >> 2;         // 0..63 : A tile row
    const int akq = (tid & 3) << 3;   // 0,8,16,24 : A tile k (8-wide)
    const int bkr = tid >> 4;         // 0..15 : B tile k row (per pass)
    const int bn4 = (tid & 15) << 2;  // 0..60 : B tile n (4-wide)

    const int  gn   = nBase + bn4;
    const bool n3ok = (gn + 3 < N);
    const float* Arow = A + (size_t)(mBase + ar) * K;

    float ra[8];   // staged A values
    float rb[8];   // staged B values (2 passes x 4)

    // ---- stage next tile's globals into registers (branch-free tails) ----
    auto loadRegs = [&](int kt) {
        const bool full = (kt + TK <= K);      // wave-uniform
        if (full) {
            const float4 v0 = *(const float4*)(Arow + kt + akq);
            const float4 v1 = *(const float4*)(Arow + kt + akq + 4);
            ra[0] = v0.x; ra[1] = v0.y; ra[2] = v0.z; ra[3] = v0.w;
            ra[4] = v1.x; ra[5] = v1.y; ra[6] = v1.z; ra[7] = v1.w;
        } else {
            #pragma unroll
            for (int i = 0; i < 8; ++i) {
                const int k  = kt + akq + i;
                const int kc = k < K ? k : K - 1;       // clamp, then mask
                ra[i] = Arow[kc] * (k < K ? 1.f : 0.f);
            }
        }
        #pragma unroll
        for (int p = 0; p < 2; ++p) {
            const int gk  = kt + p * 16 + bkr;
            const int gkc = (full || gk < K) ? gk : K - 1;
            const float km = (full || gk < K) ? 1.f : 0.f;
            const float* bp = B + (size_t)gkc * N;
            if (n3ok) {
                const float4 vv = *(const float4*)(bp + gn);
                rb[p * 4 + 0] = vv.x * km; rb[p * 4 + 1] = vv.y * km;
                rb[p * 4 + 2] = vv.z * km; rb[p * 4 + 3] = vv.w * km;
            } else {
                #pragma unroll
                for (int i = 0; i < 4; ++i) {
                    const int c  = gn + i;
                    const int cc = c < N ? c : N - 1;
                    rb[p * 4 + i] = bp[cc] * ((c < N) ? km : 0.f);
                }
            }
        }
    };

    // ---- convert staged registers to bf16 and store into LDS buffer ----
    auto storeLDS = [&](int buf) {
        unsigned short* ad = &Asm[buf][ar * AS + akq];
        #pragma unroll
        for (int i = 0; i < 4; ++i)
            *(unsigned int*)(ad + 2 * i) = pack2(ra[2 * i], ra[2 * i + 1]);
        #pragma unroll
        for (int p = 0; p < 2; ++p) {
            unsigned short* bd = &Bsm[buf][(p * 16 + bkr) * BS + bn4];
            *(unsigned int*)(bd)     = pack2(rb[p * 4 + 0], rb[p * 4 + 1]);
            *(unsigned int*)(bd + 2) = pack2(rb[p * 4 + 2], rb[p * 4 + 3]);
        }
    };

    v8f acc0 = {}, acc1 = {};
    const int nT = (K + TK - 1) / TK;

    loadRegs(0);
    storeLDS(0);
    for (int t = 0; t < nT; ++t) {
        __syncthreads();
        if (t + 1 < nT) loadRegs((t + 1) * TK);   // overlap with WMMAs below

        const int buf = t & 1;
        FragAB a, b0, b1;
        {
            const int m    = wm * 16 + (lane & 15);
            const int half = lane >> 4;
            #pragma unroll
            for (int v = 0; v < 8; ++v) {
                const int kb = ((v < 4) ? 0 : 16) + half * 8 + (v & 3) * 2;
                a.u32[v] = *(const unsigned int*)&Asm[buf][m * AS + kb];
            }
            const int nn = wn * 32 + (lane & 15);
            #pragma unroll
            for (int j = 0; j < 8; ++j) {
                const int kb = ((j < 4) ? 0 : 16) + half * 8 + (j & 3) * 2;
                b0.u32[j] = (unsigned int)Bsm[buf][kb * BS + nn] | ((unsigned int)Bsm[buf][(kb + 1) * BS + nn] << 16);
                b1.u32[j] = (unsigned int)Bsm[buf][kb * BS + nn + 16] | ((unsigned int)Bsm[buf][(kb + 1) * BS + nn + 16] << 16);
            }
        }
        acc0 = __builtin_amdgcn_wmma_f32_16x16x32_bf16(
            false, a.v, false, b0.v, (short)0, acc0, false, false);
        acc1 = __builtin_amdgcn_wmma_f32_16x16x32_bf16(
            false, a.v, false, b1.v, (short)0, acc1, false, false);

        if (t + 1 < nT) storeLDS((t + 1) & 1);
    }

    // ---- epilogue: C/D layout n=lane%16, m=r+8*(lane/16) ----
    const int half = lane >> 4;
    const int nl   = lane & 15;
    const int rowB = mBase + wm * 16 + half * 8;
    const int col0 = nBase + wn * 32 + nl;
    const int col1 = col0 + 16;
    #pragma unroll
    for (int r = 0; r < 8; ++r) {
        const int row = rowB + r;
        if (col0 < N) {
            float v = acc0[r];
            if (bias)   v += bias[col0];
            if (doRelu) v = fmaxf(v, 0.f);
            C[(size_t)row * N + col0] = v;
            if (C2) C2[(size_t)row * N + col0] = fmaxf(v, 0.f);
        }
        if (col1 < N) {
            float v = acc1[r];
            if (bias)   v += bias[col1];
            if (doRelu) v = fmaxf(v, 0.f);
            C[(size_t)row * N + col1] = v;
            if (C2) C2[(size_t)row * N + col1] = fmaxf(v, 0.f);
        }
    }
}

// ---------------------------------------------------------------------------
// Fused attention GEMM: Out[N,64] = P @ Wh, with
//   P[i,j] = adj[i,j]>0 ? exp(leaky(ssrc[i]+sdst[j]) - mrow[i]) * linv[i] : 0
// A tile is synthesized on the fly (no [N,N] materialization). N fixed 64,
// K = Nn (multiple of 32, no tails). Same double-buffered pipeline.
// ---------------------------------------------------------------------------
__global__ __launch_bounds__(256) void gemm_attn_wmma(
    const float* __restrict__ adj, const float* __restrict__ ssrc,
    const float* __restrict__ sdst, const float* __restrict__ mrow,
    const float* __restrict__ linv, const float* __restrict__ Wh,
    float* __restrict__ Out, int Nn)
{
    __shared__ unsigned short Asm[2][TM * AS];
    __shared__ unsigned short Bsm[2][TK * BS];

    const int tid  = threadIdx.x;
    const int lane = tid & 31;
    const int w    = tid >> 5;
    const int wm   = w >> 1;
    const int wn   = w & 1;
    const int mBase = blockIdx.x * TM;

    const int ar  = tid >> 2;         // 0..63
    const int akq = (tid & 3) << 3;   // 0,8,16,24
    const int bkr = tid >> 4;
    const int bn4 = (tid & 15) << 2;

    const int   gi = mBase + ar;
    const float si = ssrc[gi];
    const float mi = mrow[gi];
    const float li = linv[gi];
    const float* aprow = adj + (size_t)gi * Nn;

    float raa[8], rdd[8], rb[8];

    auto loadRegs = [&](int kt) {
        const int j0 = kt + akq;
        const float4 av0 = *(const float4*)(aprow + j0);
        const float4 av1 = *(const float4*)(aprow + j0 + 4);
        const float4 dv0 = *(const float4*)(sdst + j0);
        const float4 dv1 = *(const float4*)(sdst + j0 + 4);
        raa[0]=av0.x; raa[1]=av0.y; raa[2]=av0.z; raa[3]=av0.w;
        raa[4]=av1.x; raa[5]=av1.y; raa[6]=av1.z; raa[7]=av1.w;
        rdd[0]=dv0.x; rdd[1]=dv0.y; rdd[2]=dv0.z; rdd[3]=dv0.w;
        rdd[4]=dv1.x; rdd[5]=dv1.y; rdd[6]=dv1.z; rdd[7]=dv1.w;
        #pragma unroll
        for (int p = 0; p < 2; ++p) {
            const float4 vv = *(const float4*)(Wh + (size_t)(kt + p * 16 + bkr) * 64 + bn4);
            rb[p*4+0]=vv.x; rb[p*4+1]=vv.y; rb[p*4+2]=vv.z; rb[p*4+3]=vv.w;
        }
    };

    auto storeLDS = [&](int buf) {
        float pv[8];
        #pragma unroll
        for (int i = 0; i < 8; ++i) {
            float e = si + rdd[i];
            e = e > 0.f ? e : 0.2f * e;
            pv[i] = (raa[i] > 0.f) ? __expf(e - mi) * li : 0.f;
        }
        unsigned short* ad = &Asm[buf][ar * AS + akq];
        #pragma unroll
        for (int i = 0; i < 4; ++i)
            *(unsigned int*)(ad + 2 * i) = pack2(pv[2 * i], pv[2 * i + 1]);
        #pragma unroll
        for (int p = 0; p < 2; ++p) {
            unsigned short* bd = &Bsm[buf][(p * 16 + bkr) * BS + bn4];
            *(unsigned int*)(bd)     = pack2(rb[p * 4 + 0], rb[p * 4 + 1]);
            *(unsigned int*)(bd + 2) = pack2(rb[p * 4 + 2], rb[p * 4 + 3]);
        }
    };

    v8f acc0 = {}, acc1 = {};
    const int nT = Nn / TK;

    loadRegs(0);
    storeLDS(0);
    for (int t = 0; t < nT; ++t) {
        __syncthreads();
        if (t + 1 < nT) loadRegs((t + 1) * TK);

        const int buf = t & 1;
        FragAB a, b0, b1;
        {
            const int m    = wm * 16 + (lane & 15);
            const int half = lane >> 4;
            #pragma unroll
            for (int v = 0; v < 8; ++v) {
                const int kb = ((v < 4) ? 0 : 16) + half * 8 + (v & 3) * 2;
                a.u32[v] = *(const unsigned int*)&Asm[buf][m * AS + kb];
            }
            const int nn = wn * 32 + (lane & 15);
            #pragma unroll
            for (int j = 0; j < 8; ++j) {
                const int kb = ((j < 4) ? 0 : 16) + half * 8 + (j & 3) * 2;
                b0.u32[j] = (unsigned int)Bsm[buf][kb * BS + nn] | ((unsigned int)Bsm[buf][(kb + 1) * BS + nn] << 16);
                b1.u32[j] = (unsigned int)Bsm[buf][kb * BS + nn + 16] | ((unsigned int)Bsm[buf][(kb + 1) * BS + nn + 16] << 16);
            }
        }
        acc0 = __builtin_amdgcn_wmma_f32_16x16x32_bf16(
            false, a.v, false, b0.v, (short)0, acc0, false, false);
        acc1 = __builtin_amdgcn_wmma_f32_16x16x32_bf16(
            false, a.v, false, b1.v, (short)0, acc1, false, false);

        if (t + 1 < nT) storeLDS((t + 1) & 1);
    }

    const int half = lane >> 4;
    const int nl   = lane & 15;
    const int rowB = mBase + wm * 16 + half * 8;
    const int col0 = wn * 32 + nl;
    #pragma unroll
    for (int r = 0; r < 8; ++r) {
        Out[(size_t)(rowB + r) * 64 + col0]      = acc0[r];
        Out[(size_t)(rowB + r) * 64 + col0 + 16] = acc1[r];
    }
}

// ---------------------------------------------------------------------------
// ssrc = Wh @ a_src, sdst = Wh @ a_dst  (wave-per-row dot, F_OUT = 64)
// ---------------------------------------------------------------------------
__global__ __launch_bounds__(256) void att_scores_kernel(
    const float* __restrict__ Wh, const float* __restrict__ avs,
    const float* __restrict__ avd, float* __restrict__ ssrc,
    float* __restrict__ sdst)
{
    const int lane = threadIdx.x & 31;
    const int row  = blockIdx.x * 8 + (threadIdx.x >> 5);
    const float w0 = Wh[row * 64 + lane];
    const float w1 = Wh[row * 64 + lane + 32];
    float ps = w0 * avs[lane] + w1 * avs[lane + 32];
    float pd = w0 * avd[lane] + w1 * avd[lane + 32];
    for (int off = 16; off > 0; off >>= 1) {
        ps += __shfl_down(ps, off, 32);
        pd += __shfl_down(pd, off, 32);
    }
    if (lane == 0) { ssrc[row] = ps; sdst[row] = pd; }
}

// ---------------------------------------------------------------------------
// Per-row softmax stats: mrow[i] = max_j masked-leaky(e), linv[i] = 1/sum(exp)
// ---------------------------------------------------------------------------
__global__ __launch_bounds__(256) void att_stats_kernel(
    const float* __restrict__ adj, const float* __restrict__ ssrc,
    const float* __restrict__ sdst, float* __restrict__ mrow,
    float* __restrict__ linv, int Nn)
{
    __shared__ float red[256];
    const int i   = blockIdx.x;
    const int tid = threadIdx.x;
    const float si = ssrc[i];
    float ev[16];
    float mx = -3.4e38f;
    #pragma unroll
    for (int p = 0; p < 16; ++p) {
        const int j = p * 256 + tid;
        const float a = adj[(size_t)i * Nn + j];
        float e = si + sdst[j];
        e = e > 0.f ? e : 0.2f * e;
        e = (a > 0.f) ? e : -3.0e38f;
        ev[p] = e;
        mx = fmaxf(mx, e);
    }
    red[tid] = mx; __syncthreads();
    for (int s = 128; s > 0; s >>= 1) {
        if (tid < s) red[tid] = fmaxf(red[tid], red[tid + s]);
        __syncthreads();
    }
    mx = red[0]; __syncthreads();
    float sum = 0.f;
    #pragma unroll
    for (int p = 0; p < 16; ++p) sum += __expf(ev[p] - mx);
    red[tid] = sum; __syncthreads();
    for (int s = 128; s > 0; s >>= 1) {
        if (tid < s) red[tid] += red[tid + s];
        __syncthreads();
    }
    if (tid == 0) { mrow[i] = mx; linv[i] = 1.f / red[0]; }
}

// ---------------------------------------------------------------------------
// rowsum of graph_neigh (block per row)
// ---------------------------------------------------------------------------
__global__ __launch_bounds__(256) void rowsum_kernel(
    const float* __restrict__ A, float* __restrict__ out, int Nn)
{
    __shared__ float red[256];
    const int i   = blockIdx.x;
    const int tid = threadIdx.x;
    float s = 0.f;
    for (int p = 0; p < 16; ++p) s += A[(size_t)i * Nn + p * 256 + tid];
    red[tid] = s; __syncthreads();
    for (int st = 128; st > 0; st >>= 1) {
        if (tid < st) red[tid] += red[tid + st];
        __syncthreads();
    }
    if (tid == 0) out[i] = red[0];
}

// ---------------------------------------------------------------------------
// readout: g = sigmoid( (gpre/rowsum) / max(||.||2, 1e-12) ), F_OUT = 64
// ---------------------------------------------------------------------------
__global__ __launch_bounds__(64) void readout_norm_kernel(
    const float* __restrict__ gpre, const float* __restrict__ rs,
    float* __restrict__ g)
{
    __shared__ float red[64];
    const int i = blockIdx.x, t = threadIdx.x;
    const float v = gpre[(size_t)i * 64 + t] / rs[i];
    red[t] = v * v; __syncthreads();
    for (int s = 32; s > 0; s >>= 1) {
        if (t < s) red[t] += red[t + s];
        __syncthreads();
    }
    const float d = fmaxf(sqrtf(red[0]), 1e-12f);
    const float x = v / d;
    g[(size_t)i * 64 + t] = 1.f / (1.f + __expf(-x));
}

// ---------------------------------------------------------------------------
// bilinear row-dot: out[row*stride+offs] = dot(X[row,:64], Y[row,:64]) + b
// ---------------------------------------------------------------------------
__global__ __launch_bounds__(256) void rowdot_kernel(
    const float* __restrict__ X, const float* __restrict__ Y,
    const float* __restrict__ bptr, float* __restrict__ outp,
    int stride, int offs)
{
    const int lane = threadIdx.x & 31;
    const int row  = blockIdx.x * 8 + (threadIdx.x >> 5);
    float p = X[row * 64 + lane] * Y[row * 64 + lane]
            + X[row * 64 + lane + 32] * Y[row * 64 + lane + 32];
    for (int off = 16; off > 0; off >>= 1) p += __shfl_down(p, off, 32);
    if (lane == 0) outp[(size_t)row * stride + offs] = p + bptr[0];
}

// ---------------------------------------------------------------------------
// Host orchestration
// ---------------------------------------------------------------------------
extern "C" void kernel_launch(void* const* d_in, const int* in_sizes, int n_in,
                              void* d_out, int out_size, void* d_ws, size_t ws_size,
                              hipStream_t stream) {
    (void)in_sizes; (void)n_in; (void)out_size; (void)ws_size;
    const float* feat     = (const float*)d_in[0];
    const float* feat_a   = (const float*)d_in[1];
    const float* adj      = (const float*)d_in[2];
    const float* gneigh   = (const float*)d_in[3];
    const float* weight1  = (const float*)d_in[4];
    const float* weight2  = (const float*)d_in[5];
    const float* att_W    = (const float*)d_in[6];
    const float* a_src    = (const float*)d_in[7];
    const float* a_dst    = (const float*)d_in[8];
    const float* mlp_W1   = (const float*)d_in[9];
    const float* mlp_b1   = (const float*)d_in[10];
    const float* mlp_W2   = (const float*)d_in[11];
    const float* mlp_b2   = (const float*)d_in[12];
    const float* mlp_W3   = (const float*)d_in[13];
    const float* mlp_b3   = (const float*)d_in[14];
    const float* dmlp_W1  = (const float*)d_in[15];
    const float* dmlp_b1  = (const float*)d_in[16];
    const float* dmlp_W2  = (const float*)d_in[17];
    const float* dmlp_b2  = (const float*)d_in[18];
    const float* bil_W    = (const float*)d_in[19];
    const float* bil_b    = (const float*)d_in[20];

    float* outp = (float*)d_out;
    float* ws   = (float*)d_ws;

    const int Nn = 4096, FIN = 3000, F = 64;
    const size_t NF = (size_t)Nn * F;           // 262144

    size_t o = 0;
    auto alloc = [&](size_t n) { float* p = ws + o; o += n; return p; };
    float* h1    = alloc(NF);
    float* h1a   = alloc(NF);
    float* Wh    = alloc(NF);
    float* Wha   = alloc(NF);
    float* att   = alloc(NF);
    float* atta  = alloc(NF);
    float* za    = alloc(NF);
    float* emb   = alloc(NF);
    float* emba  = alloc(NF);
    float* az    = alloc(NF);
    float* gpre  = alloc(NF);
    float* gprea = alloc(NF);
    float* gbuf  = alloc(NF);
    float* gabuf = alloc(NF);
    float* Demb  = alloc(NF);
    float* Demba = alloc(NF);
    float* Dg    = alloc(NF);
    float* Dga   = alloc(NF);
    float* VE    = alloc(NF);
    float* VA    = alloc(NF);
    float* t256  = alloc((size_t)Nn * 256);
    float* t128  = alloc((size_t)Nn * 128);
    float* ssrc  = alloc(Nn);
    float* sdst  = alloc(Nn);
    float* ssrca = alloc(Nn);
    float* sdsta = alloc(Nn);
    float* mrow  = alloc(Nn);
    float* linv  = alloc(Nn);
    float* mrowa = alloc(Nn);
    float* linva = alloc(Nn);
    float* rsum  = alloc(Nn);

    float* z_out  = outp;                        // hiden_emb: [4096,64]
    float* h_out  = outp + NF;                   // h: [4096,3000]
    float* ret    = h_out + (size_t)Nn * FIN;    // [4096,2]
    float* ret_a  = ret + (size_t)Nn * 2;        // [4096,2]

    const dim3 blk(256);
    auto gemm = [&](const float* A, const float* B, const float* bias,
                    float* C, float* C2, int M, int N, int K, int relu) {
        dim3 grid(M / TM, (N + TN - 1) / TN);
        gemm_bf16_wmma<<<grid, blk, 0, stream>>>(A, B, bias, C, C2, M, N, K, relu);
    };

    // --- encoder front: h = feat @ weight1, then GAT attention ---
    gemm(feat,   weight1, nullptr, h1,  nullptr, Nn, F, FIN, 0);
    gemm(feat_a, weight1, nullptr, h1a, nullptr, Nn, F, FIN, 0);
    gemm(h1,  att_W, nullptr, Wh,  nullptr, Nn, F, F, 0);
    gemm(h1a, att_W, nullptr, Wha, nullptr, Nn, F, F, 0);

    att_scores_kernel<<<Nn / 8, blk, 0, stream>>>(Wh,  a_src, a_dst, ssrc,  sdst);
    att_scores_kernel<<<Nn / 8, blk, 0, stream>>>(Wha, a_src, a_dst, ssrca, sdsta);
    att_stats_kernel<<<Nn, blk, 0, stream>>>(adj, ssrc,  sdst,  mrow,  linv,  Nn);
    att_stats_kernel<<<Nn, blk, 0, stream>>>(adj, ssrca, sdsta, mrowa, linva, Nn);
    gemm_attn_wmma<<<dim3(Nn / TM), blk, 0, stream>>>(adj, ssrc,  sdst,  mrow,  linv,  Wh,  att,  Nn);
    gemm_attn_wmma<<<dim3(Nn / TM), blk, 0, stream>>>(adj, ssrca, sdsta, mrowa, linva, Wha, atta, Nn);

    // --- MLP head: z (-> d_out) and emb = relu(z) in one pass ---
    gemm(att,  mlp_W1, mlp_b1, t256, nullptr, Nn, 256, F,   1);
    gemm(t256, mlp_W2, mlp_b2, t128, nullptr, Nn, 128, 256, 1);
    gemm(t128, mlp_W3, mlp_b3, z_out, emb,    Nn, F,   128, 0);
    gemm(atta, mlp_W1, mlp_b1, t256, nullptr, Nn, 256, F,   1);
    gemm(t256, mlp_W2, mlp_b2, t128, nullptr, Nn, 128, 256, 1);
    gemm(t128, mlp_W3, mlp_b3, za,   emba,    Nn, F,   128, 0);

    // --- h = adj @ (z @ weight2) == (adj @ z) @ weight2  (27x FLOP cut) ---
    gemm(adj, z_out,   nullptr, az,    nullptr, Nn, F,   Nn, 0);
    gemm(az,  weight2, nullptr, h_out, nullptr, Nn, FIN, F,  0);

    // --- readout: g = sigmoid(l2norm(graph_neigh@emb / rowsum)) ---
    gemm(gneigh, emb,  nullptr, gpre,  nullptr, Nn, F, Nn, 0);
    gemm(gneigh, emba, nullptr, gprea, nullptr, Nn, F, Nn, 0);
    rowsum_kernel<<<Nn, blk, 0, stream>>>(gneigh, rsum, Nn);
    readout_norm_kernel<<<Nn, dim3(64), 0, stream>>>(gpre,  rsum, gbuf);
    readout_norm_kernel<<<Nn, dim3(64), 0, stream>>>(gprea, rsum, gabuf);

    // --- discriminator MLP on c, h_pl, h_mi ---
    auto dmlp = [&](const float* x, float* y) {
        gemm(x,    dmlp_W1, dmlp_b1, t128, nullptr, Nn, 128, F,   1);
        gemm(t128, dmlp_W2, dmlp_b2, y,    nullptr, Nn, F,   128, 0);
    };
    dmlp(gbuf,  Dg);
    dmlp(gabuf, Dga);
    dmlp(emb,   Demb);
    dmlp(emba,  Demba);

    // --- bilinear: einsum('ni,ij,nj->n') = rowdot(x@W, y) + b ---
    gemm(Demb,  bil_W, nullptr, VE, nullptr, Nn, F, F, 0);
    gemm(Demba, bil_W, nullptr, VA, nullptr, Nn, F, F, 0);
    rowdot_kernel<<<Nn / 8, blk, 0, stream>>>(VE, Dg,  bil_b, ret,   2, 0);
    rowdot_kernel<<<Nn / 8, blk, 0, stream>>>(VA, Dg,  bil_b, ret,   2, 1);
    rowdot_kernel<<<Nn / 8, blk, 0, stream>>>(VA, Dga, bil_b, ret_a, 2, 0);
    rowdot_kernel<<<Nn / 8, blk, 0, stream>>>(VE, Dga, bil_b, ret_a, 2, 1);
}
